// PAE_61907658604560
// MI455X (gfx1250) — matrix operand order repaired and checked by reading, not compiled, not run on hardware
//
#include <hip/hip_runtime.h>
#include <hip/hip_bf16.h>

// ---------------------------------------------------------------------------
// PAE forward, fused for gfx1250 (MI455X).
//   prep:   f32 weights -> bf16 GEMM layouts in ws (k-major, channel-minor K)
//   stage1: conv1(WMMA) -> LN -> ELU -> conv2(WMMA) -> DFT/fc/sinusoid -> sig
//   stage2: deconv1(WMMA) -> LN -> ELU -> deconv2 last column (WMMA) -> out
// All matrix math: v_wmma_f32_16x16x32_bf16, f32 accumulation.
// conv1 B-fragments address the padded input tile LINEARLY (row stride ==
// channel count, so LDS offset == im2col K index). conv2/deconv2 pad channels
// to 256 so decode is shift/mask only. LDS rows carry +16 element physical
// padding to break power-of-two bank strides. Loops stay rolled (unroll<=2)
// to keep VGPR pressure under the spill threshold seen with full unroll.
// ---------------------------------------------------------------------------

typedef __bf16 bf16_t;
typedef __attribute__((ext_vector_type(16))) __bf16 v16bf;
typedef __attribute__((ext_vector_type(8)))  __bf16 v8bf;
typedef __attribute__((ext_vector_type(8)))  float  v8f;

union Frag { v16bf f; v8bf h[2]; };

__device__ __forceinline__ v8bf zero8() {
  v8bf z;
#pragma unroll
  for (int i = 0; i < 8; ++i) z[i] = (__bf16)0.0f;
  return z;
}
__device__ __forceinline__ v8f zero8f() {
  v8f z;
#pragma unroll
  for (int i = 0; i < 8; ++i) z[i] = 0.0f;
  return z;
}

#define TPI 6.28318530717958647692f
#define RS  272                      // padded LDS row stride (256 logical + 16)

// ws byte offsets (all 256-aligned)
#define WB1_OFF 0u         // [240][736]  bf16  (K' = k*48 + cin, pad 720->736)
#define WB2_OFF 353280u    // [16][3840]  bf16  (K' = k*256 + c; rows 6..15 zero)
#define WB3_OFF 476160u    // [240][128]  bf16  (K' = k*8 + e,  pad -> 128)
#define WB4_OFF 537600u    // [48][2048]  bf16  (K' = k*256 + c, k=0..7)
#define SIG_OFF 734208u    // [B][15][8]  bf16

// ---------------------------------------------------------------------------
__global__ void pae_prep_weights(const float* __restrict__ w1,
                                 const float* __restrict__ w2,
                                 const float* __restrict__ w3,
                                 const float* __restrict__ w4,
                                 bf16_t* __restrict__ wb1, bf16_t* __restrict__ wb2,
                                 bf16_t* __restrict__ wb3, bf16_t* __restrict__ wb4) {
  const int tid = blockIdx.x * blockDim.x + threadIdx.x;
  const int nth = gridDim.x * blockDim.x;
  for (int i = tid; i < 240 * 736; i += nth) {           // conv1_w [240][48][15]
    const int co = i / 736, kp = i % 736;
    const int k = kp / 48, ci = kp % 48;
    wb1[i] = (bf16_t)((kp < 720) ? w1[co * 720 + ci * 15 + k] : 0.0f);
  }
  for (int i = tid; i < 16 * 3840; i += nth) {           // conv2_w [6][240][15], rows 6..15 zero
    const int e = i / 3840, kp = i % 3840;
    const int k = kp >> 8, c = kp & 255;
    wb2[i] = (bf16_t)((e < 6 && c < 240) ? w2[e * 3600 + c * 15 + k] : 0.0f);
  }
  for (int i = tid; i < 240 * 128; i += nth) {           // deconv1_w [240][6][15]
    const int co = i >> 7, kp = i & 127;
    const int k = kp >> 3, e = kp & 7;
    wb3[i] = (bf16_t)((e < 6 && k < 15) ? w3[co * 90 + e * 15 + k] : 0.0f);
  }
  for (int i = tid; i < 48 * 2048; i += nth) {           // deconv2_w [48][240][15], taps k=0..7
    const int oc = i >> 11, kp = i & 2047;
    const int k = kp >> 8, c = kp & 255;
    wb4[i] = (bf16_t)((c < 240) ? w4[oc * 3600 + c * 15 + k] : 0.0f);
  }
}

// ---------------------------------------------------------------------------
// Stage 1: 16 samples per workgroup, 8 waves.
__launch_bounds__(256, 1)
__global__ void pae_stage1(const float* __restrict__ x,
                           const bf16_t* __restrict__ wb1,
                           const bf16_t* __restrict__ wb2,
                           const float* __restrict__ conv1_b,
                           const float* __restrict__ norm1_w,
                           const float* __restrict__ norm1_b,
                           const float* __restrict__ conv2_b,
                           const float* __restrict__ fc_w,
                           const float* __restrict__ fc_b,
                           bf16_t* __restrict__ sigg) {
  __shared__ __align__(16) char   s_ab[166912];   // xs[16][32][48] + w1l[80][736] ; later w2l[16][3856]
  __shared__ __align__(16) bf16_t s_h1[240 * RS]; // [sample*15+s][272] bf16, 240 logical ch + pad
  __shared__ float s_h2[16 * 6 * 15];             // [sample][e][s]
  __shared__ float s_b1[240];
  __shared__ float s_sum[16], s_sq[16], s_mu[16], s_rs[16];

  bf16_t* xs  = (bf16_t*)s_ab;                    // 24576 elems
  bf16_t* w1l = (bf16_t*)(s_ab + 49152);          // 58880 elems
  bf16_t* w2l = (bf16_t*)s_ab;                    // 16*3856 elems (aliases xs+w1l after conv1)

  const int tid  = threadIdx.x;
  const int lane = tid & 31;
  const int wv   = tid >> 5;
  const int b0   = blockIdx.x * 16;
  const int hi16 = (lane & 16) ? 1 : 0;

  // ---- stage x into LDS, bf16, time-padded rows [0..6]=0, [7..21]=data, [22..31]=0
  for (int i = tid; i < 16 * 32 * 48; i += 256) xs[i] = (bf16_t)0.0f;
  for (int i = tid; i < 240; i += 256) s_b1[i] = conv1_b[i];
  // zero the channel-pad columns of s_h1 (c = 240..255); conv1 never writes them
  for (int i = tid; i < 240 * 16; i += 256) s_h1[(i >> 4) * RS + 240 + (i & 15)] = (bf16_t)0.0f;
  __syncthreads();
  for (int i = tid; i < 16 * 720; i += 256) {
    const int sm = i / 720, o = i % 720;
    const int ci = o / 15, s = o % 15;
    xs[(sm * 32 + (s + 7)) * 48 + ci] = (bf16_t)x[(b0 + sm) * 720 + o];
  }
  __syncthreads();

  // ---- conv1: A = weights (M=cout, 3 chunks of 80), B = activations (N=(b,s))
  for (int ch = 0; ch < 3; ++ch) {
    const int cbase = ch * 80;
    for (int i = tid * 8; i < 80 * 736; i += 256 * 8)
      *(v8bf*)&w1l[i] = *(const v8bf*)&wb1[cbase * 736 + i];
    __syncthreads();

    for (int nt = wv; nt < 15; nt += 8) {
      const int col = nt * 16 + (lane & 15);
      const int sm = col / 15, s = col % 15;
      // LDS offset of the im2col patch is LINEAR in K': addr = xoff + K'
      const int xoff  = (sm * 32 + s) * 48 + hi16 * 16;
      const int wbase = (lane & 15) * 736 + hi16 * 16;
      v8f acc[5];
#pragma unroll
      for (int m = 0; m < 5; ++m) acc[m] = zero8f();

#pragma unroll 2
      for (int kk = 0; kk < 22; ++kk) {             // K' = 0..703, never masked
        Frag bfr;
        const bf16_t* p = &xs[xoff + kk * 32];
        bfr.h[0] = *(const v8bf*)p;
        bfr.h[1] = *(const v8bf*)(p + 8);
        const int ao = wbase + kk * 32;
#pragma unroll
        for (int m = 0; m < 5; ++m) {
          Frag af;
          const bf16_t* wp = &w1l[m * (16 * 736) + ao];
          af.h[0] = *(const v8bf*)wp;
          af.h[1] = *(const v8bf*)(wp + 8);
          acc[m] = __builtin_amdgcn_wmma_f32_16x16x32_bf16(false, af.f, false, bfr.f,
                                                           (short)0, acc[m], false, false);
        }
      }
      { // kk = 22: hi-lane half (K' 720..735) is zero padding
        Frag bfr;
        if (!hi16) {
          const bf16_t* p = &xs[xoff + 22 * 32];
          bfr.h[0] = *(const v8bf*)p;
          bfr.h[1] = *(const v8bf*)(p + 8);
        } else { bfr.h[0] = zero8(); bfr.h[1] = zero8(); }
        const int ao = wbase + 22 * 32;
#pragma unroll
        for (int m = 0; m < 5; ++m) {
          Frag af;
          const bf16_t* wp = &w1l[m * (16 * 736) + ao];
          af.h[0] = *(const v8bf*)wp;
          af.h[1] = *(const v8bf*)(wp + 8);
          acc[m] = __builtin_amdgcn_wmma_f32_16x16x32_bf16(false, af.f, false, bfr.f,
                                                           (short)0, acc[m], false, false);
        }
      }
#pragma unroll
      for (int m = 0; m < 5; ++m) {
#pragma unroll
        for (int r = 0; r < 8; ++r) {
          const int co = cbase + m * 16 + r + hi16 * 8;
          s_h1[(sm * 15 + s) * RS + co] = (bf16_t)(acc[m][r] + s_b1[co]);
        }
      }
    }
    __syncthreads();
  }

  // ---- layernorm over 3600 per sample + ELU (in place)
  if (tid < 16) { s_sum[tid] = 0.0f; s_sq[tid] = 0.0f; }
  __syncthreads();
  {
    const int sm = tid >> 4, pt = tid & 15;
    float ls = 0.0f, lq = 0.0f;
    for (int rr = 0; rr < 15; ++rr)
      for (int c = pt; c < 240; c += 16) {
        const float v = (float)s_h1[(sm * 15 + rr) * RS + c];
        ls += v; lq += v * v;
      }
    atomicAdd(&s_sum[sm], ls);
    atomicAdd(&s_sq[sm], lq);
  }
  __syncthreads();
  if (tid < 16) {
    const float mu = s_sum[tid] * (1.0f / 3600.0f);
    const float var = s_sq[tid] * (1.0f / 3600.0f) - mu * mu;
    s_mu[tid] = mu;
    s_rs[tid] = rsqrtf(var + 1e-5f);
  }
  __syncthreads();
  for (int i = tid; i < 16 * 3600; i += 256) {
    const int sm = i / 3600, o = i % 3600;
    const int s = o / 240, c = o % 240;
    const int jn = c * 15 + s;                         // reference norm index
    const int a = (sm * 15 + s) * RS + c;
    float v = ((float)s_h1[a] - s_mu[sm]) * s_rs[sm] * norm1_w[jn] + norm1_b[jn];
    v = (v > 0.0f) ? v : expm1f(v);
    s_h1[a] = (bf16_t)v;
  }
  __syncthreads();

  // ---- conv2 weights -> LDS (16 rows incl. zero rows 6..15), re-stride 3840->3856
  for (int i = tid * 8; i < 16 * 3840; i += 256 * 8) {
    const int e = i / 3840, kp = i % 3840;
    *(v8bf*)&w2l[e * 3856 + kp] = *(const v8bf*)&wb2[i];
  }
  __syncthreads();

  // ---- conv2: A = activations (M=(b,s)), B = weights (N=16, 6 real embs)
  //      K = 15 taps * 256 padded channels = 3840 -> 120 steps, shift-decode
  for (int mt = wv; mt < 15; mt += 8) {
    const int row = mt * 16 + (lane & 15);
    const int sm = row / 15, s = row % 15;
    const int e = lane & 15;
    v8f acc = zero8f();
#pragma unroll 2
    for (int kk = 0; kk < 120; ++kk) {
      Frag af, bfr;
#pragma unroll
      for (int hh = 0; hh < 2; ++hh) {
        const int idx = kk * 32 + hi16 * 16 + hh * 8;  // K' = k*256 + c
        const int k = idx >> 8, c0 = idx & 255;
        const int si = s + k - 7;
        v8bf run = zero8();
        if (si >= 0 && si < 15)
          run = *(const v8bf*)&s_h1[(sm * 15 + si) * RS + c0];
        af.h[hh] = run;
      }
      const bf16_t* p = &w2l[e * 3856 + kk * 32 + hi16 * 16];
      bfr.h[0] = *(const v8bf*)p;
      bfr.h[1] = *(const v8bf*)(p + 8);
      acc = __builtin_amdgcn_wmma_f32_16x16x32_bf16(false, af.f, false, bfr.f,
                                                    (short)0, acc, false, false);
    }
#pragma unroll
    for (int r = 0; r < 8; ++r) {
      if (e < 6) {
        const int row2 = mt * 16 + r + hi16 * 8;
        const int sm2 = row2 / 15, s2 = row2 % 15;
        s_h2[(sm2 * 6 + e) * 15 + s2] = acc[r] + conv2_b[e];
      }
    }
  }
  __syncthreads();

  // ---- DFT bins 1..7, amplitude / freq / offset, fc->atan2 phase, sinusoid
  if (tid < 96) {
    const int sm = tid / 6, e = tid % 6;
    float h[15];
    float off = 0.0f;
#pragma unroll
    for (int s = 0; s < 15; ++s) { h[s] = s_h2[(sm * 6 + e) * 15 + s]; off += h[s]; }
    off *= (1.0f / 15.0f);
    float psum = 0.0f, fsum = 0.0f;
#pragma unroll
    for (int n = 1; n <= 7; ++n) {
      float re = 0.0f, im = 0.0f;
#pragma unroll
      for (int s = 0; s < 15; ++s) {
        const float ang = TPI * (float)(n * s) * (1.0f / 15.0f);
        re += h[s] * __cosf(ang);
        im -= h[s] * __sinf(ang);
      }
      const float pw = re * re + im * im;
      psum += pw;
      fsum += ((float)n * (1.0f / 0.3f)) * pw;        // freqs[n] = n / WINDOW
    }
    const float f = fsum / psum;
    const float a = 2.0f * sqrtf(psum) * (1.0f / 15.0f);
    float v0 = fc_b[e * 2 + 0], v1 = fc_b[e * 2 + 1];
#pragma unroll
    for (int s = 0; s < 15; ++s) {
      v0 += h[s] * fc_w[(e * 2 + 0) * 15 + s];
      v1 += h[s] * fc_w[(e * 2 + 1) * 15 + s];
    }
    const float p = atan2f(v1, v0) * (1.0f / TPI);
#pragma unroll
    for (int s = 0; s < 15; ++s) {
      const float arg = -0.15f + (float)s * (0.3f / 14.0f);
      const float sg = a * __sinf(TPI * (f * arg + p)) + off;
      sigg[((b0 + sm) * 15 + s) * 8 + e] = (bf16_t)sg;
    }
  } else if (tid < 128) {                              // zero padding lanes e=6,7
    const int t2 = tid - 96;
    const int sm = t2 >> 1, e = 6 + (t2 & 1);
    for (int s = 0; s < 15; ++s)
      sigg[((b0 + sm) * 15 + s) * 8 + e] = (bf16_t)0.0f;
  }
}

// ---------------------------------------------------------------------------
// Stage 2: deconv1 -> LN -> ELU -> deconv2 (last time step only)
__launch_bounds__(256, 1)
__global__ void pae_stage2(const bf16_t* __restrict__ sigg,
                           const bf16_t* __restrict__ wb3,
                           const bf16_t* __restrict__ wb4,
                           const float* __restrict__ deconv1_b,
                           const float* __restrict__ denorm1_w,
                           const float* __restrict__ denorm1_b,
                           const float* __restrict__ deconv2_b,
                           float* __restrict__ out) {
  __shared__ __align__(16) bf16_t s_sig[16 * 15 * 8];
  __shared__ __align__(16) bf16_t s_w3[240 * 136];     // rows padded 128 -> 136
  __shared__ __align__(16) bf16_t s_h3[240 * RS];      // [sample*15+s][272]
  __shared__ float s_b3[240];
  __shared__ float s_sum[16], s_sq[16], s_mu[16], s_rs[16];

  const int tid  = threadIdx.x;
  const int lane = tid & 31;
  const int wv   = tid >> 5;
  const int b0   = blockIdx.x * 16;
  const int hi16 = (lane & 16) ? 1 : 0;

  for (int i = tid * 8; i < 16 * 15 * 8; i += 256 * 8)
    *(v8bf*)&s_sig[i] = *(const v8bf*)&sigg[b0 * 120 + i];
  for (int i = tid * 8; i < 240 * 128; i += 256 * 8) { // re-stride 128 -> 136
    const int row = i >> 7, kp = i & 127;
    *(v8bf*)&s_w3[row * 136 + kp] = *(const v8bf*)&wb3[i];
  }
  for (int i = tid; i < 240; i += 256) s_b3[i] = deconv1_b[i];
  for (int i = tid; i < 240 * 16; i += 256) s_h3[(i >> 4) * RS + 240 + (i & 15)] = (bf16_t)0.0f;
  __syncthreads();

  // ---- deconv1: A = sig patches (M=(b,s)), B = weights (N=cout), K=128
  for (int mt = 0; mt < 15; ++mt) {
    const int row = mt * 16 + (lane & 15);
    const int sm = row / 15, s = row % 15;
    Frag af[4];
#pragma unroll
    for (int kk = 0; kk < 4; ++kk) {
#pragma unroll
      for (int hh = 0; hh < 2; ++hh) {
        const int idx = kk * 32 + hi16 * 16 + hh * 8;  // K' = k*8 + e
        const int k = idx >> 3;
        const int si = s + k - 7;
        v8bf run = zero8();
        if (k < 15 && si >= 0 && si < 15)
          run = *(const v8bf*)&s_sig[(sm * 15 + si) * 8];
        af[kk].h[hh] = run;
      }
    }
    for (int nt = wv; nt < 15; nt += 8) {
      const int co = nt * 16 + (lane & 15);
      v8f acc = zero8f();
#pragma unroll
      for (int kk = 0; kk < 4; ++kk) {
        Frag bfr;
        const bf16_t* p = &s_w3[co * 136 + kk * 32 + hi16 * 16];
        bfr.h[0] = *(const v8bf*)p;
        bfr.h[1] = *(const v8bf*)(p + 8);
        acc = __builtin_amdgcn_wmma_f32_16x16x32_bf16(false, af[kk].f, false, bfr.f,
                                                      (short)0, acc, false, false);
      }
#pragma unroll
      for (int r = 0; r < 8; ++r) {
        const int row2 = mt * 16 + r + hi16 * 8;
        const int sm2 = row2 / 15, s2 = row2 % 15;
        s_h3[(sm2 * 15 + s2) * RS + co] = (bf16_t)(acc[r] + s_b3[co]);
      }
    }
  }
  __syncthreads();

  // ---- layernorm2 + ELU (in place)
  if (tid < 16) { s_sum[tid] = 0.0f; s_sq[tid] = 0.0f; }
  __syncthreads();
  {
    const int sm = tid >> 4, pt = tid & 15;
    float ls = 0.0f, lq = 0.0f;
    for (int rr = 0; rr < 15; ++rr)
      for (int c = pt; c < 240; c += 16) {
        const float v = (float)s_h3[(sm * 15 + rr) * RS + c];
        ls += v; lq += v * v;
      }
    atomicAdd(&s_sum[sm], ls);
    atomicAdd(&s_sq[sm], lq);
  }
  __syncthreads();
  if (tid < 16) {
    const float mu = s_sum[tid] * (1.0f / 3600.0f);
    const float var = s_sq[tid] * (1.0f / 3600.0f) - mu * mu;
    s_mu[tid] = mu;
    s_rs[tid] = rsqrtf(var + 1e-5f);
  }
  __syncthreads();
  for (int i = tid; i < 16 * 3600; i += 256) {
    const int sm = i / 3600, o = i % 3600;
    const int s = o / 240, c = o % 240;
    const int jn = c * 15 + s;
    const int a = (sm * 15 + s) * RS + c;
    float v = ((float)s_h3[a] - s_mu[sm]) * s_rs[sm] * denorm1_w[jn] + denorm1_b[jn];
    v = (v > 0.0f) ? v : expm1f(v);
    s_h3[a] = (bf16_t)v;
  }
  __syncthreads();

  // ---- deconv2 last column: M = 16 samples (1 tile), N = 48 oc (3 tiles),
  //      K = 8 taps * 256 padded channels = 2048 -> 64 steps, shift-decode
  if (wv < 3) {
    const int sm = lane & 15;
    const int oc = wv * 16 + (lane & 15);
    v8f acc = zero8f();
#pragma unroll 2
    for (int kk = 0; kk < 64; ++kk) {
      Frag af, bfr;
#pragma unroll
      for (int hh = 0; hh < 2; ++hh) {
        const int idx = kk * 32 + hi16 * 16 + hh * 8;  // K' = k*256 + c, k=0..7 -> s=7+k
        const int k = idx >> 8, c0 = idx & 255;
        af.h[hh] = *(const v8bf*)&s_h3[(sm * 15 + (7 + k)) * RS + c0];
      }
      const bf16_t* p = &wb4[oc * 2048 + kk * 32 + hi16 * 16];
      bfr.h[0] = *(const v8bf*)p;
      bfr.h[1] = *(const v8bf*)(p + 8);
      acc = __builtin_amdgcn_wmma_f32_16x16x32_bf16(false, af.f, false, bfr.f,
                                                    (short)0, acc, false, false);
    }
#pragma unroll
    for (int r = 0; r < 8; ++r) {
      const int sm2 = r + hi16 * 8;
      out[(b0 + sm2) * 48 + oc] = acc[r] + deconv2_b[oc];
    }
  }
}

// ---------------------------------------------------------------------------
extern "C" void kernel_launch(void* const* d_in, const int* in_sizes, int n_in,
                              void* d_out, int out_size, void* d_ws, size_t ws_size,
                              hipStream_t stream) {
  (void)in_sizes; (void)n_in; (void)out_size; (void)ws_size;
  const float* x         = (const float*)d_in[0];
  const float* conv1_w   = (const float*)d_in[1];
  const float* conv1_b   = (const float*)d_in[2];
  const float* norm1_w   = (const float*)d_in[3];
  const float* norm1_b   = (const float*)d_in[4];
  const float* conv2_w   = (const float*)d_in[5];
  const float* conv2_b   = (const float*)d_in[6];
  const float* fc_w      = (const float*)d_in[7];
  const float* fc_b      = (const float*)d_in[8];
  const float* deconv1_w = (const float*)d_in[9];
  const float* deconv1_b = (const float*)d_in[10];
  const float* denorm1_w = (const float*)d_in[11];
  const float* denorm1_b = (const float*)d_in[12];
  const float* deconv2_w = (const float*)d_in[13];
  const float* deconv2_b = (const float*)d_in[14];
  float* out = (float*)d_out;

  char* ws = (char*)d_ws;
  bf16_t* wb1 = (bf16_t*)(ws + WB1_OFF);
  bf16_t* wb2 = (bf16_t*)(ws + WB2_OFF);
  bf16_t* wb3 = (bf16_t*)(ws + WB3_OFF);
  bf16_t* wb4 = (bf16_t*)(ws + WB4_OFF);
  bf16_t* sig = (bf16_t*)(ws + SIG_OFF);

  pae_prep_weights<<<512, 256, 0, stream>>>(conv1_w, conv2_w, deconv1_w, deconv2_w,
                                            wb1, wb2, wb3, wb4);
  pae_stage1<<<1024, 256, 0, stream>>>(x, wb1, wb2, conv1_b, norm1_w, norm1_b,
                                       conv2_b, fc_w, fc_b, sig);
  pae_stage2<<<1024, 256, 0, stream>>>(sig, wb3, wb4, deconv1_b, denorm1_w,
                                       denorm1_b, deconv2_b, out);
}
